// RNN_56573309224307
// MI455X (gfx1250) — matrix operand order repaired and checked by reading, not compiled, not run on hardware
//
#include <hip/hip_runtime.h>
#include <hip/hip_bf16.h>

// ---------------------------------------------------------------------------
// RNN (tanh, 4 layers) + linear decoder for MI455X (gfx1250, wave32, WMMA).
// bf16 A/B fragments -> V_WMMA_F32_16X16X32_BF16 (f32 accumulate).
// Recurrence runs as a persistent kernel (1 launch/layer) with a device-scope
// barrier between timesteps; W_hh tiles live in LDS for all 512 steps.
// ---------------------------------------------------------------------------

typedef __attribute__((ext_vector_type(16))) __bf16 bf16x16;
typedef __attribute__((ext_vector_type(8)))  __bf16 bf16x8;
typedef __attribute__((ext_vector_type(8)))  float  floatx8;

#define DEV_INLINE __device__ __forceinline__

static constexpr int T_  = 512;
static constexpr int B_  = 64;
static constexpr int IN_ = 1024;
static constexpr int H_  = 1024;
static constexpr int L_  = 4;
static constexpr int C_  = 50;
static constexpr int TB_ = T_ * B_;          // 32768 rows in time-parallel GEMMs
static constexpr int SCAN_BLOCKS = 32;       // persistent-scan grid (co-resident)

// ---------------------------------------------------------------------------
// fp32 -> bf16 (round-to-nearest-even) bulk converter
// ---------------------------------------------------------------------------
__global__ void f32_to_bf16_kernel(const float* __restrict__ src,
                                   unsigned short* __restrict__ dst, int n) {
  int i = blockIdx.x * blockDim.x + threadIdx.x;
  if (i < n) {
    unsigned int u = __float_as_uint(src[i]);
    unsigned int r = (u + 0x7FFFu + ((u >> 16) & 1u)) >> 16;
    dst[i] = (unsigned short)r;
  }
}

__global__ void init_ctr_kernel(unsigned int* __restrict__ ctr, int n) {
  int i = blockIdx.x * blockDim.x + threadIdx.x;
  if (i < n) ctr[i] = 0u;
}

DEV_INLINE unsigned short f32_to_bf16(float v) {
  unsigned int u = __float_as_uint(v);
  return (unsigned short)((u + 0x7FFFu + ((u >> 16) & 1u)) >> 16);
}

// fast tanh: (e^{2x}-1)/(e^{2x}+1), clamped so exp stays finite
DEV_INLINE float tanh_fast(float x) {
  float xc = fminf(fmaxf(x, -15.0f), 15.0f);
  float e  = __expf(2.0f * xc);
  return (e - 1.0f) / (e + 1.0f);
}

// 16x32 (MxK) bf16 A fragment, contiguous K order:
// lane<16: K = k0+0..15; lane>=16: K = k0+16..31.
DEV_INLINE bf16x16 load_a_frag(const unsigned short* __restrict__ rowptr, int k0) {
  bf16x8 lo = *(const bf16x8*)((const __bf16*)rowptr + k0);
  bf16x8 hi = *(const bf16x8*)((const __bf16*)rowptr + k0 + 8);
  bf16x16 a;
#pragma unroll
  for (int i = 0; i < 8; ++i) { a[i] = lo[i]; a[i + 8] = hi[i]; }
  return a;
}

DEV_INLINE bf16x16 wmma_bf16(bf16x16 a, bf16x16 b);  // (unused decl guard)

// ---------------------------------------------------------------------------
// XP = X(bf16)[TB,K] * W(bf16)[N,K]^T + (bih+bhh)  -> fp32 [TB,N]
// 2x2 register blocking: each wave computes a 32x32 tile (4 WMMA / 8 loads).
// Block = 8 waves (2 M x 4 N) covering 64(M) x 128(N). grid = (TB/64, N/128).
// ---------------------------------------------------------------------------
__global__ __launch_bounds__(256) void xp_gemm_kernel(
    const unsigned short* __restrict__ X,
    const unsigned short* __restrict__ W,
    const float* __restrict__ bih, const float* __restrict__ bhh,
    float* __restrict__ XP) {
  constexpr int K = IN_, N = H_;
  const int wave = threadIdx.x >> 5;
  const int lane = threadIdx.x & 31;
  const int m0 = blockIdx.x * 64 + (wave >> 2) * 32;
  const int n0 = blockIdx.y * 128 + (wave & 3) * 32;
  const int row = lane & 15;
  const int hi  = lane >> 4;

  const unsigned short* A0 = X + (size_t)(m0 + row) * K + hi * 16;
  const unsigned short* A1 = A0 + (size_t)16 * K;
  const unsigned short* B0 = W + (size_t)(n0 + row) * K + hi * 16;
  const unsigned short* B1 = B0 + (size_t)16 * K;

  floatx8 c00 = {}, c01 = {}, c10 = {}, c11 = {};
#pragma unroll 2
  for (int k0 = 0; k0 < K; k0 += 32) {
    bf16x16 a0 = load_a_frag(A0, k0);
    bf16x16 a1 = load_a_frag(A1, k0);
    bf16x16 b0 = *(const bf16x16*)((const __bf16*)B0 + k0);
    bf16x16 b1 = *(const bf16x16*)((const __bf16*)B1 + k0);
    c00 = __builtin_amdgcn_wmma_f32_16x16x32_bf16(false, a0, false, b0, (short)0, c00, false, false);
    c01 = __builtin_amdgcn_wmma_f32_16x16x32_bf16(false, a0, false, b1, (short)0, c01, false, false);
    c10 = __builtin_amdgcn_wmma_f32_16x16x32_bf16(false, a1, false, b0, (short)0, c10, false, false);
    c11 = __builtin_amdgcn_wmma_f32_16x16x32_bf16(false, a1, false, b1, (short)0, c11, false, false);
  }

  const int col0 = n0 + row, col1 = n0 + 16 + row;
  const float bias0 = bih[col0] + bhh[col0];
  const float bias1 = bih[col1] + bhh[col1];
  float* o00 = XP + (size_t)(m0 + hi * 8) * N + col0;
  float* o10 = o00 + (size_t)16 * N;
#pragma unroll
  for (int r = 0; r < 8; ++r) {
    o00[(size_t)r * N]      = c00[r] + bias0;
    o00[(size_t)r * N + 16] = c01[r] + bias1;
    o10[(size_t)r * N]      = c10[r] + bias0;
    o10[(size_t)r * N + 16] = c11[r] + bias1;
  }
}

// ---------------------------------------------------------------------------
// Persistent recurrence: for t in 0..T-1: y[t] = tanh(XP[t] + y[t-1]*Whh^T).
// Grid = 32 blocks x 8 waves; block owns 2 N-tiles x all 4 M-tiles.
// The block's 32 Whh rows (64KB bf16) are staged in LDS once and feed
// ds_load B fragments for all 512 steps. Steps are separated by a
// device-scope atomic release/acquire barrier (all 32 blocks co-resident).
// ---------------------------------------------------------------------------
__global__ __launch_bounds__(256) void rnn_scan_kernel(
    const unsigned short* __restrict__ h0,    // [B,H] bf16 initial hidden
    const unsigned short* __restrict__ Whh,   // [H,H] bf16 row-major
    const float* __restrict__ XP,             // [T,B,H] f32
    unsigned short* __restrict__ Y,           // [T,B,H] bf16 outputs
    float* __restrict__ HidF32,               // [B,H] f32 final hidden
    unsigned int* __restrict__ ctr)           // zeroed barrier counter
{
  constexpr int K = H_, N = H_;
  __shared__ __attribute__((aligned(32))) unsigned short ldsW[32 * H_];  // 64KB

  const int wave = threadIdx.x >> 5;
  const int lane = threadIdx.x & 31;
  const int mt   = wave & 3;                  // batch tile 0..3
  const int ntl  = wave >> 2;                 // local N tile 0..1
  const int nBase = blockIdx.x * 32;          // this block's 32 Whh rows
  const int m0 = mt * 16;
  const int row = lane & 15;
  const int hi  = lane >> 4;

  // stage Whh rows [nBase, nBase+32) into LDS (flat 64KB copy)
  {
    const uint4* src = (const uint4*)(Whh + (size_t)nBase * K);
    uint4* dst = (uint4*)ldsW;
    for (int i = threadIdx.x; i < 32 * K / 8; i += 256) dst[i] = src[i];
  }
  __syncthreads();

  const unsigned short* ldsRow = ldsW + (size_t)(ntl * 16 + row) * K + hi * 16;
  const int col = nBase + ntl * 16 + row;
  const size_t cbase = (size_t)(m0 + hi * 8) * N + col;

  for (int t = 0; t < T_; ++t) {
    const unsigned short* Hin = (t == 0) ? h0 : (Y + (size_t)(t - 1) * B_ * H_);
    const unsigned short* Arow = Hin + (size_t)(m0 + row) * K + hi * 16;
    const float* XPt = XP + (size_t)t * B_ * H_;
    unsigned short* Yt = Y + (size_t)t * B_ * H_;

    floatx8 c = {};
#pragma unroll 4
    for (int k0 = 0; k0 < K; k0 += 32) {
      bf16x16 a = load_a_frag(Arow, k0);
      bf16x16 b = *(const bf16x16*)((const __bf16*)ldsRow + k0);  // ds_load
      c = __builtin_amdgcn_wmma_f32_16x16x32_bf16(false, a, false, b,
                                                  (short)0, c, false, false);
    }
#pragma unroll
    for (int r = 0; r < 8; ++r) {
      float v = tanh_fast(c[r] + XPt[cbase + (size_t)r * N]);
      Yt[cbase + (size_t)r * N] = f32_to_bf16(v);
      if (t == T_ - 1) HidF32[cbase + (size_t)r * N] = v;
    }

    // ---- device-wide barrier between timesteps ----
    __threadfence();                          // release this wave's stores
    __syncthreads();                          // whole block done + fenced
    if (threadIdx.x == 0) {
      __hip_atomic_fetch_add(ctr, 1u, __ATOMIC_RELEASE, __HIP_MEMORY_SCOPE_AGENT);
      const unsigned int target = (unsigned int)gridDim.x * (unsigned int)(t + 1);
      while (__hip_atomic_load(ctr, __ATOMIC_ACQUIRE, __HIP_MEMORY_SCOPE_AGENT) < target)
        __builtin_amdgcn_s_sleep(1);
    }
    __syncthreads();                          // block waits on the spinner
  }
}

// ---------------------------------------------------------------------------
// decoded[m,c] = Y(bf16)[m,:] . Wdec(bf16)[c,:] + bdec[c];  N padded 50 -> 64
// block = 8 waves = 2 M-tiles x 4 N-tiles; grid = TB/32
// ---------------------------------------------------------------------------
__global__ __launch_bounds__(256) void decode_kernel(
    const unsigned short* __restrict__ Y,
    const unsigned short* __restrict__ Wd,
    const float* __restrict__ bdec,
    float* __restrict__ out) {
  constexpr int K = H_;
  const int wave = threadIdx.x >> 5;
  const int lane = threadIdx.x & 31;
  const int m0 = (blockIdx.x * 2 + (wave >> 2)) * 16;
  const int n0 = (wave & 3) * 16;
  const int row = lane & 15;
  const int hi  = lane >> 4;

  const int bcol = n0 + row;
  const int bclamp = (bcol < C_) ? bcol : (C_ - 1);   // safe load, masked store
  const unsigned short* Arow = Y  + (size_t)(m0 + row) * K + hi * 16;
  const unsigned short* Brow = Wd + (size_t)bclamp   * K + hi * 16;

  floatx8 c = {};
#pragma unroll 4
  for (int k0 = 0; k0 < K; k0 += 32) {
    bf16x16 a = load_a_frag(Arow, k0);
    bf16x16 b = *(const bf16x16*)((const __bf16*)Brow + k0);
    c = __builtin_amdgcn_wmma_f32_16x16x32_bf16(false, a, false, b,
                                                (short)0, c, false, false);
  }
  if (bcol < C_) {
    const float bias = bdec[bcol];
    float* o = out + (size_t)(m0 + hi * 8) * C_ + bcol;
#pragma unroll
    for (int r = 0; r < 8; ++r) o[(size_t)r * C_] = c[r] + bias;
  }
}

// ---------------------------------------------------------------------------
extern "C" void kernel_launch(void* const* d_in, const int* in_sizes, int n_in,
                              void* d_out, int out_size, void* d_ws, size_t ws_size,
                              hipStream_t stream) {
  (void)in_sizes; (void)n_in; (void)out_size; (void)ws_size;
  const float* in_x = (const float*)d_in[0];   // [T,B,IN]
  const float* h0   = (const float*)d_in[1];   // [L,B,H]
  const float* Wih  = (const float*)d_in[2];   // [L,H,IN]
  const float* Whh  = (const float*)d_in[3];   // [L,H,H]
  const float* bih  = (const float*)d_in[4];   // [L,H]
  const float* bhh  = (const float*)d_in[5];   // [L,H]
  const float* Wdec = (const float*)d_in[6];   // [C,H]
  const float* bdec = (const float*)d_in[7];   // [C]
  float* out = (float*)d_out;                  // decoded [T,B,C] then hidden [L,B,H]

  // -------- workspace carve (bf16 stored as ushort) --------
  char* p = (char*)d_ws;
  auto take = [&](size_t bytes) { char* r = p; p += (bytes + 255) & ~(size_t)255; return r; };
  unsigned short* xbf   = (unsigned short*)take((size_t)TB_ * H_ * 2);  // layer input acts
  unsigned short* ybf   = (unsigned short*)take((size_t)TB_ * H_ * 2);  // layer output acts
  float*          xp    = (float*)         take((size_t)TB_ * H_ * 4); // input projection
  unsigned short* wihb  = (unsigned short*)take((size_t)L_ * H_ * IN_ * 2);
  unsigned short* whhb  = (unsigned short*)take((size_t)L_ * H_ * H_  * 2);
  unsigned short* wdecb = (unsigned short*)take((size_t)C_ * H_ * 2);
  unsigned short* h0b   = (unsigned short*)take((size_t)L_ * B_ * H_ * 2);
  unsigned int*   ctrs  = (unsigned int*)  take((size_t)L_ * sizeof(unsigned int));

  init_ctr_kernel<<<1, 32, 0, stream>>>(ctrs, L_);

  auto cvt = [&](const float* s, unsigned short* d, int n) {
    f32_to_bf16_kernel<<<(n + 255) / 256, 256, 0, stream>>>(s, d, n);
  };
  cvt(in_x, xbf,   TB_ * IN_);
  cvt(h0,   h0b,   L_ * B_ * H_);
  cvt(Wih,  wihb,  L_ * H_ * IN_);
  cvt(Whh,  whhb,  L_ * H_ * H_);
  cvt(Wdec, wdecb, C_ * H_);

  float* hid_out = out + (size_t)TB_ * C_;     // [L,B,H] section of d_out

  unsigned short* xcur = xbf;
  unsigned short* ycur = ybf;
  for (int l = 0; l < L_; ++l) {
    // time-parallel input projection (+ both biases), fp32 result
    xp_gemm_kernel<<<dim3(TB_ / 64, H_ / 128), 256, 0, stream>>>(
        xcur, wihb + (size_t)l * H_ * IN_, bih + l * H_, bhh + l * H_, xp);

    // one persistent kernel runs the whole 512-step scan for this layer
    rnn_scan_kernel<<<SCAN_BLOCKS, 256, 0, stream>>>(
        h0b + (size_t)l * B_ * H_, whhb + (size_t)l * H_ * H_, xp,
        ycur, hid_out + (size_t)l * B_ * H_, ctrs + l);

    // next layer consumes this layer's outputs
    unsigned short* tmp = xcur; xcur = ycur; ycur = tmp;
  }

  // decoder on the last layer's outputs (now in xcur after the swap)
  decode_kernel<<<TB_ / 32, 256, 0, stream>>>(xcur, wdecb, bdec, out);
}
